// OpinionDynamicsModel_3959959847022
// MI455X (gfx1250) — matrix operand order repaired and checked by reading, not compiled, not run on hardware
//
#include <hip/hip_runtime.h>

// ---------------------------------------------------------------------------
// Types for CDNA5 WMMA
// ---------------------------------------------------------------------------
typedef __attribute__((ext_vector_type(16))) __bf16 v16bf;
typedef __attribute__((ext_vector_type(8)))  __bf16 v8bf;
typedef __attribute__((ext_vector_type(8)))  float  v8f;

// 16-byte vector type matching the async-to-LDS builtin's pointee type
typedef int v4i_ __attribute__((vector_size(16)));
typedef __attribute__((address_space(1))) v4i_* gas_v4i_ptr;
typedef __attribute__((address_space(3))) v4i_* las_v4i_ptr;

#define D_IN  768
#define D_HID 128
#define HEADS 4

// ---------------------------------------------------------------------------
// gfx1250 async global->LDS helpers (ASYNCcnt-tracked DMA)
// ---------------------------------------------------------------------------
__device__ __forceinline__ void async_load_b128_to_lds(const __bf16* g, __bf16* l) {
#if defined(__has_builtin) && __has_builtin(__builtin_amdgcn_global_load_async_to_lds_b128)
    __builtin_amdgcn_global_load_async_to_lds_b128(
        (gas_v4i_ptr)g,
        (las_v4i_ptr)l,
        0, 0);
#else
    unsigned loff = (unsigned)(size_t)(__attribute__((address_space(3))) void*)l;
    asm volatile("global_load_async_to_lds_b128 %0, %1, off"
                 :: "v"(loff), "v"(g) : "memory");
#endif
}

__device__ __forceinline__ void wait_asynccnt0() {
#if defined(__has_builtin) && __has_builtin(__builtin_amdgcn_s_wait_asynccnt)
    __builtin_amdgcn_s_wait_asynccnt(0);
#else
    asm volatile("s_wait_asynccnt 0x0" ::: "memory");
#endif
}

// ---------------------------------------------------------------------------
// Utility kernels
// ---------------------------------------------------------------------------
__global__ void fill_u32_kernel(unsigned int* p, unsigned int v, long n) {
    long i = (long)blockIdx.x * blockDim.x + threadIdx.x;
    if (i < n) p[i] = v;
}

__global__ void cvt_f32_bf16_kernel(const float* __restrict__ s, __bf16* __restrict__ d, long n) {
    long i = (long)blockIdx.x * blockDim.x + threadIdx.x;
    if (i < n) d[i] = (__bf16)s[i];
}

// W is [K, Nc] row-major (fp32). Bt is [Nc, K] row-major (bf16): Bt[n*K+k] = W[k*Nc+n]
__global__ void transpose_cvt_kernel(const float* __restrict__ W, __bf16* __restrict__ Bt,
                                     int K, int Nc) {
    long i = (long)blockIdx.x * blockDim.x + threadIdx.x;
    long total = (long)K * Nc;
    if (i >= total) return;
    int n = (int)(i / K);
    int k = (int)(i - (long)n * K);
    Bt[i] = (__bf16)W[(long)k * Nc + n];
}

// ---------------------------------------------------------------------------
// WMMA GEMM:  C[M,Nc](f32) = A[M,K](bf16, row-major) * Bt[Nc,K](bf16)^T + bias
// block = 256 threads = 8 waves; block tile = 128 rows x 128 cols;
// wave tile = 16 rows x 128 cols (8 accumulators of 16x16), K-step = 32.
// B panel (128 cols x 32 k = 8KB) is staged in LDS via async-to-LDS DMA,
// double buffered; LDS column stride = 40 elements (80B) -> 16B aligned and
// bank-conflict-free ds_load_b128 fragment reads.
// ---------------------------------------------------------------------------
#define BSTRIDE 40   // __bf16 elements per LDS column (32 data + 8 pad)

__global__ __launch_bounds__(256)
void gemm_bf16_wmma_kernel(const __bf16* __restrict__ A, const __bf16* __restrict__ Bt,
                           const float* __restrict__ bias, float* __restrict__ C,
                           int M, int K, int Nc) {
    __shared__ __bf16 ldsB[2][128 * BSTRIDE];

    const int tid  = threadIdx.x;
    const int lane = tid & 31;
    const int wv   = tid >> 5;
    const int l15  = lane & 15;
    const int lhi  = lane >> 4;          // 0 for lanes 0-15, 1 for lanes 16-31

    const int r0     = blockIdx.x * 128 + wv * 16;
    const int c_base = blockIdx.y * 128;

    // --- async staging: thread t stages half a column (16 elements = 32B) ---
    const int scol  = tid >> 1;          // 0..127 (local column)
    const int shalf = tid & 1;           // which 16-element half of the 32-k chunk
    const __bf16* gB = Bt + (size_t)(c_base + scol) * K + shalf * 16;
    __bf16* lB[2] = { &ldsB[0][scol * BSTRIDE + shalf * 16],
                      &ldsB[1][scol * BSTRIDE + shalf * 16] };

    // --- A fragment addressing (ISA 16-bit A 16x32 layout) ---
    int rowA = r0 + l15;
    if (rowA >= M) rowA = M - 1;         // clamp (stores are masked)
    const __bf16* arow = A + (size_t)rowA * K + (lhi << 3);

    v8f acc[8] = {};

    // prologue: stage panel for k0 = 0 into buffer 0
    async_load_b128_to_lds(gB, lB[0]);
    async_load_b128_to_lds(gB + 8, lB[0] + 8);
    wait_asynccnt0();
    __syncthreads();

    int p = 0;
    for (int k0 = 0; k0 < K; k0 += 32) {
        const bool more = (k0 + 32) < K;
        if (more) {   // overlap DMA of next panel with this panel's WMMAs
            const __bf16* gsrc = gB + k0 + 32;
            async_load_b128_to_lds(gsrc, lB[p ^ 1]);
            async_load_b128_to_lds(gsrc + 8, lB[p ^ 1] + 8);
            __builtin_prefetch(arow + k0 + 32, 0, 3);
        }

        // A fragment: two 16B chunks at k0+{0,16} (lanes 16-31 shifted by 8)
        v8bf alo = *(const v8bf*)(arow + k0);
        v8bf ahi = *(const v8bf*)(arow + k0 + 16);
        v16bf af = __builtin_shufflevector(alo, ahi, 0, 1, 2, 3, 4, 5, 6, 7,
                                                     8, 9, 10, 11, 12, 13, 14, 15);

        // batch-read all 8 B fragments from LDS, then issue the WMMA chain
        const __bf16* lbase = &ldsB[p][lhi * 8];
        v16bf bfv[8];
#pragma unroll
        for (int ct = 0; ct < 8; ++ct) {
            const __bf16* bl = lbase + (ct * 16 + l15) * BSTRIDE;
            v8bf blo = *(const v8bf*)bl;
            v8bf bhi = *(const v8bf*)(bl + 16);
            bfv[ct] = __builtin_shufflevector(blo, bhi, 0, 1, 2, 3, 4, 5, 6, 7,
                                                        8, 9, 10, 11, 12, 13, 14, 15);
        }
#pragma unroll
        for (int ct = 0; ct < 8; ++ct)
            acc[ct] = __builtin_amdgcn_wmma_f32_16x16x32_bf16(
                false, af, false, bfv[ct], (short)0, acc[ct], false, false);

        if (more) {
            wait_asynccnt0();   // next panel landed
            __syncthreads();    // all waves done reading current panel
            p ^= 1;
        }
    }

    // Store: C/D 16x16 layout — VGPR v: lanes 0-15 -> row r0+v, lanes 16-31 -> r0+8+v;
    // column = lane&15 for every VGPR of the tile.
    const int rbase = r0 + (lhi << 3);
#pragma unroll
    for (int ct = 0; ct < 8; ++ct) {
        const int col = c_base + ct * 16 + l15;
        const float bv = bias ? bias[col] : 0.0f;
#pragma unroll
        for (int v = 0; v < 8; ++v) {
            const int row = rbase + v;
            if (row < M) C[(size_t)row * Nc + col] = acc[ct][v] + bv;
        }
    }
}

// ---------------------------------------------------------------------------
// Attention scores: a_src[n,h] = dot(h[n, h*128 : h*128+128], att_src[h]), same for dst.
// One wave per (node, head); block = H*32 threads.
// ---------------------------------------------------------------------------
__global__ void att_scores_kernel(const float* __restrict__ h,
                                  const float* __restrict__ atts,
                                  const float* __restrict__ attd,
                                  float* __restrict__ as_, float* __restrict__ ad_,
                                  int N, int H) {
    int n = blockIdx.x;
    if (n >= N) return;
    int hd = threadIdx.x >> 5, lane = threadIdx.x & 31;
    const float* hp = h + (size_t)n * H * 128 + (size_t)hd * 128;
    const float* sp = atts + (size_t)hd * 128;
    const float* dp = attd + (size_t)hd * 128;
    float ss = 0.f, dd = 0.f;
#pragma unroll
    for (int c = lane; c < 128; c += 32) {
        float v = hp[c];
        ss += v * sp[c];
        dd += v * dp[c];
    }
    for (int o = 16; o > 0; o >>= 1) {
        ss += __shfl_down(ss, o);
        dd += __shfl_down(dd, o);
    }
    if (lane == 0) {
        as_[(size_t)n * H + hd] = ss;
        ad_[(size_t)n * H + hd] = dd;
    }
}

__device__ __forceinline__ unsigned int enc_f32(float v) {
    unsigned int u = __float_as_uint(v);
    return (u & 0x80000000u) ? ~u : (u | 0x80000000u);
}
__device__ __forceinline__ float dec_f32(unsigned int u) {
    return (u & 0x80000000u) ? __uint_as_float(u & 0x7fffffffu) : __uint_as_float(~u);
}

// Pass A: e = leaky_relu(a_src[src]+a_dst[dst]); segment-max at dst via ordered-uint atomicMax
__global__ void edge_passA_kernel(const int* __restrict__ ei, int E, int ET, int H,
                                  const float* __restrict__ as_, const float* __restrict__ ad_,
                                  float* __restrict__ eval, unsigned int* __restrict__ menc) {
    long idx = (long)blockIdx.x * blockDim.x + threadIdx.x;
    long total = (long)ET * H;
    if (idx >= total) return;
    int e = (int)(idx / H), h = (int)(idx - (long)e * H);
    int s = (e < E) ? ei[e] : (e - E);
    int d = (e < E) ? ei[E + e] : (e - E);
    float v = as_[(size_t)s * H + h] + ad_[(size_t)d * H + h];
    v = v > 0.f ? v : 0.2f * v;
    eval[idx] = v;
    atomicMax(&menc[(size_t)d * H + h], enc_f32(v));
}

// Pass B: ex = exp(e - m[dst]) (in place); segment-sum at dst
__global__ void edge_passB_kernel(const int* __restrict__ ei, int E, int ET, int H,
                                  float* __restrict__ eval,
                                  const unsigned int* __restrict__ menc,
                                  float* __restrict__ ssum) {
    long idx = (long)blockIdx.x * blockDim.x + threadIdx.x;
    long total = (long)ET * H;
    if (idx >= total) return;
    int e = (int)(idx / H), h = (int)(idx - (long)e * H);
    int d = (e < E) ? ei[E + e] : (e - E);
    float m = dec_f32(menc[(size_t)d * H + h]);
    float x = expf(eval[idx] - m);
    eval[idx] = x;
    atomicAdd(&ssum[(size_t)d * H + h], x);
}

// Pass C: agg[dst,h,:] += (ex / (s[dst,h]+eps)) * h[src,h,:] ; one wave per (edge, head)
__global__ void edge_passC_kernel(const int* __restrict__ ei, int E, int ET, int H,
                                  const float* __restrict__ eval,
                                  const float* __restrict__ ssum,
                                  const float* __restrict__ hfeat,
                                  float* __restrict__ agg) {
    long gtid = (long)blockIdx.x * blockDim.x + threadIdx.x;
    long wid = gtid >> 5;
    int lane = (int)(gtid & 31);
    long total = (long)ET * H;
    if (wid >= total) return;
    int e = (int)(wid / H), h = (int)(wid - (long)e * H);
    int s = (e < E) ? ei[e] : (e - E);
    int d = (e < E) ? ei[E + e] : (e - E);
    float alpha = eval[wid] / (ssum[(size_t)d * H + h] + 1e-16f);
    const float* hs = hfeat + (size_t)s * H * 128 + (size_t)h * 128;
    float* ag = agg + (size_t)d * H * 128 + (size_t)h * 128;
#pragma unroll
    for (int c = lane; c < 128; c += 32)
        atomicAdd(&ag[c], alpha * hs[c]);
}

// Epilogue gat1: out = elu(agg + bias) -> bf16 (input to next GEMM). ld = 512.
__global__ void epilogue1_kernel(const float* __restrict__ agg, const float* __restrict__ bias,
                                 __bf16* __restrict__ out, long total) {
    long i = (long)blockIdx.x * blockDim.x + threadIdx.x;
    if (i >= total) return;
    float v = agg[i] + bias[i & 511];
    v = v > 0.f ? v : (expf(v) - 1.0f);
    out[i] = (__bf16)v;
}

// Epilogue gat2 + residual: out = (agg + bias + beta*gp) -> bf16. ld = 128.
__global__ void epilogue2_kernel(const float* __restrict__ agg, const float* __restrict__ bias,
                                 const float* __restrict__ gp, const float* __restrict__ beta,
                                 __bf16* __restrict__ out, long total) {
    long i = (long)blockIdx.x * blockDim.x + threadIdx.x;
    if (i >= total) return;
    float v = agg[i] + bias[i & 127] + beta[0] * gp[i];
    out[i] = (__bf16)v;
}

// ---------------------------------------------------------------------------
// Host-side launcher
// ---------------------------------------------------------------------------
static inline long cdiv(long a, long b) { return (a + b - 1) / b; }

extern "C" void kernel_launch(void* const* d_in, const int* in_sizes, int n_in,
                              void* d_out, int out_size, void* d_ws, size_t ws_size,
                              hipStream_t stream) {
    const float* x        = (const float*)d_in[0];
    const int*   ei       = (const int*)d_in[1];
    const float* g        = (const float*)d_in[2];
    const float* proj_W   = (const float*)d_in[3];
    const float* proj_b   = (const float*)d_in[4];
    const float* gat1_W   = (const float*)d_in[5];
    const float* g1_asrc  = (const float*)d_in[6];
    const float* g1_adst  = (const float*)d_in[7];
    const float* g1_bias  = (const float*)d_in[8];
    const float* gat2_W   = (const float*)d_in[9];
    const float* g2_asrc  = (const float*)d_in[10];
    const float* g2_adst  = (const float*)d_in[11];
    const float* g2_bias  = (const float*)d_in[12];
    const float* dec_W    = (const float*)d_in[13];
    const float* dec_b    = (const float*)d_in[14];
    const float* beta     = (const float*)d_in[15];
    float* out            = (float*)d_out;

    const int N  = in_sizes[0] / D_IN;   // 10000
    const int E  = in_sizes[1] / 2;      // 160000
    const int ET = E + N;                // edges + self loops

    // --- workspace arena (256B-aligned bump allocator) ---
    char* base = (char*)d_ws;
    size_t off = 0;
    auto carve = [&](size_t bytes) -> void* {
        void* p = base + off;
        off = (off + bytes + 255) & ~(size_t)255;
        return p;
    };
    __bf16* actb   = (__bf16*)carve((size_t)N * D_IN * 2);     // shared bf16 activation buf
    __bf16* projWt = (__bf16*)carve((size_t)D_HID * D_IN * 2);            // [128,768]
    __bf16* gat1Wt = (__bf16*)carve((size_t)(HEADS * D_HID) * D_HID * 2); // [512,128]
    __bf16* gat2Wt = (__bf16*)carve((size_t)D_HID * (HEADS * D_HID) * 2); // [128,512]
    __bf16* decWt  = (__bf16*)carve((size_t)D_IN * D_HID * 2);            // [768,128]
    float*  xp     = (float*)carve((size_t)N * D_HID * 4);     // reused as h2 later
    float*  gp     = (float*)carve((size_t)N * D_HID * 4);
    float*  h1     = (float*)carve((size_t)N * HEADS * D_HID * 4);
    float*  agg    = (float*)carve((size_t)N * HEADS * D_HID * 4); // agg1, later agg2
    float*  as1    = (float*)carve((size_t)N * HEADS * 4);
    float*  ad1    = (float*)carve((size_t)N * HEADS * 4);
    unsigned int* m1 = (unsigned int*)carve((size_t)N * HEADS * 4);
    float*  s1     = (float*)carve((size_t)N * HEADS * 4);
    float*  e1     = (float*)carve((size_t)ET * HEADS * 4);
    float*  as2    = (float*)carve((size_t)N * 4);
    float*  ad2    = (float*)carve((size_t)N * 4);
    unsigned int* m2 = (unsigned int*)carve((size_t)N * 4);
    float*  s2     = (float*)carve((size_t)N * 4);
    float*  e2     = (float*)carve((size_t)ET * 4);
    float*  h2     = xp; // xp dead after its bf16 conversion

    const int TB = 256;
    auto gemm = [&](const __bf16* A, const __bf16* Bt, const float* bias, float* C,
                    int M, int K, int Nc) {
        dim3 grid((unsigned)cdiv(M, 128), (unsigned)(Nc / 128));
        gemm_bf16_wmma_kernel<<<grid, TB, 0, stream>>>(A, Bt, bias, C, M, K, Nc);
    };

    // 0) weight transpose+convert (fp32 -> bf16, [K,N] -> [N,K])
    transpose_cvt_kernel<<<cdiv((long)D_IN * D_HID, TB), TB, 0, stream>>>(proj_W, projWt, D_IN, D_HID);
    transpose_cvt_kernel<<<cdiv((long)D_HID * HEADS * D_HID, TB), TB, 0, stream>>>(gat1_W, gat1Wt, D_HID, HEADS * D_HID);
    transpose_cvt_kernel<<<cdiv((long)HEADS * D_HID * D_HID, TB), TB, 0, stream>>>(gat2_W, gat2Wt, HEADS * D_HID, D_HID);
    transpose_cvt_kernel<<<cdiv((long)D_HID * D_IN, TB), TB, 0, stream>>>(dec_W, decWt, D_HID, D_IN);

    // 1) xp = x @ proj_W + proj_b
    cvt_f32_bf16_kernel<<<cdiv((long)N * D_IN, TB), TB, 0, stream>>>(x, actb, (long)N * D_IN);
    gemm(actb, projWt, proj_b, xp, N, D_IN, D_HID);

    // 2) gp = g @ proj_W + proj_b
    cvt_f32_bf16_kernel<<<cdiv((long)N * D_IN, TB), TB, 0, stream>>>(g, actb, (long)N * D_IN);
    gemm(actb, projWt, proj_b, gp, N, D_IN, D_HID);

    // 3) h1 = xp @ gat1_W   [N, 512]
    cvt_f32_bf16_kernel<<<cdiv((long)N * D_HID, TB), TB, 0, stream>>>(xp, actb, (long)N * D_HID);
    gemm(actb, gat1Wt, nullptr, h1, N, D_HID, HEADS * D_HID);

    // 4) gat1 attention + aggregation
    att_scores_kernel<<<N, HEADS * 32, 0, stream>>>(h1, g1_asrc, g1_adst, as1, ad1, N, HEADS);
    fill_u32_kernel<<<cdiv((long)N * HEADS, TB), TB, 0, stream>>>(m1, 0u, (long)N * HEADS);
    fill_u32_kernel<<<cdiv((long)N * HEADS, TB), TB, 0, stream>>>((unsigned int*)s1, 0u, (long)N * HEADS);
    edge_passA_kernel<<<cdiv((long)ET * HEADS, TB), TB, 0, stream>>>(ei, E, ET, HEADS, as1, ad1, e1, m1);
    edge_passB_kernel<<<cdiv((long)ET * HEADS, TB), TB, 0, stream>>>(ei, E, ET, HEADS, e1, m1, s1);
    fill_u32_kernel<<<cdiv((long)N * HEADS * D_HID, TB), TB, 0, stream>>>((unsigned int*)agg, 0u, (long)N * HEADS * D_HID);
    edge_passC_kernel<<<cdiv((long)ET * HEADS * 32, TB), TB, 0, stream>>>(ei, E, ET, HEADS, e1, s1, h1, agg);

    // 5) elu(agg + bias) -> bf16; h2 = elu_out @ gat2_W
    epilogue1_kernel<<<cdiv((long)N * HEADS * D_HID, TB), TB, 0, stream>>>(agg, g1_bias, actb, (long)N * HEADS * D_HID);
    gemm(actb, gat2Wt, nullptr, h2, N, HEADS * D_HID, D_HID);

    // 6) gat2 attention + aggregation (H=1)
    att_scores_kernel<<<N, 32, 0, stream>>>(h2, g2_asrc, g2_adst, as2, ad2, N, 1);
    fill_u32_kernel<<<cdiv((long)N, TB), TB, 0, stream>>>(m2, 0u, (long)N);
    fill_u32_kernel<<<cdiv((long)N, TB), TB, 0, stream>>>((unsigned int*)s2, 0u, (long)N);
    edge_passA_kernel<<<cdiv((long)ET, TB), TB, 0, stream>>>(ei, E, ET, 1, as2, ad2, e2, m2);
    edge_passB_kernel<<<cdiv((long)ET, TB), TB, 0, stream>>>(ei, E, ET, 1, e2, m2, s2);
    fill_u32_kernel<<<cdiv((long)N * D_HID, TB), TB, 0, stream>>>((unsigned int*)agg, 0u, (long)N * D_HID);
    edge_passC_kernel<<<cdiv((long)ET * 32, TB), TB, 0, stream>>>(ei, E, ET, 1, e2, s2, h2, agg);

    // 7) combine: (agg + bias + beta*gp) -> bf16; decode: out = hc @ dec_W + dec_b
    epilogue2_kernel<<<cdiv((long)N * D_HID, TB), TB, 0, stream>>>(agg, g2_bias, gp, beta, actb, (long)N * D_HID);
    gemm(actb, decWt, dec_b, out, N, D_HID, D_IN);
}
